// HGAT_4784593568241
// MI455X (gfx1250) — matrix operand order repaired and checked by reading, not compiled, not run on hardware
//
#include <hip/hip_runtime.h>
#include <cstdint>

#define NNODES 50000
#define NEDGES 500000

typedef __attribute__((ext_vector_type(16))) __bf16 v16bf;
typedef __attribute__((ext_vector_type(8)))  __bf16 v8bf;
typedef __attribute__((ext_vector_type(8)))  float  v8f;

// ---------------- weight convert into transposed+padded bf16 [Npad][Kpad] ----
// trans==0: W stored [Nw][K]  (fc/nt/lin style, [out,in])  -> direct copy
// trans==1: W stored [K][Nw]  (gW/resW style, [in,out])    -> transpose
__global__ void wconv_kernel(const float* __restrict__ W, __bf16* __restrict__ out,
                             int K, int Nw, int Kpad, int Npad, int trans) {
  int i = blockIdx.x * blockDim.x + threadIdx.x;
  if (i >= Kpad * Npad) return;
  int n = i / Kpad, k = i - n * Kpad;
  float v = 0.f;
  if (n < Nw && k < K) v = trans ? W[(size_t)k * Nw + n] : W[(size_t)n * K + k];
  out[i] = (__bf16)v;
}

// ---------------- bf16 WMMA GEMM ----------------
// C[M,Nw] = A[M,Ktot](f32, lda) * Bt^T + bias ; Bt is bf16 [Npad][Kpad] (ldb=Kpad)
// Requires: grid.y*64 <= Npad, Kpad % 32 == 0, B zero-padded for k >= Ktot.
// KTAIL=false requires Ktot % 32 == 0. A rows are clamped (never masked);
// out-of-range rows compute garbage that is never stored.
#define BM 128
#define BN 64
#define BK 32
#define LDSK 40 // BK + 8 pad -> 80B row stride, conflict-free b128 LDS loads

__device__ __forceinline__ v8bf pack8(const float4& a, const float4& b) {
  v8bf p;
  p[0] = (__bf16)a.x; p[1] = (__bf16)a.y; p[2] = (__bf16)a.z; p[3] = (__bf16)a.w;
  p[4] = (__bf16)b.x; p[5] = (__bf16)b.y; p[6] = (__bf16)b.z; p[7] = (__bf16)b.w;
  return p;
}

template <bool KTAIL>
__global__ __launch_bounds__(256, 2)
void gemm_bf16_wmma(const float* __restrict__ A, int lda,
                    const __bf16* __restrict__ Bt, int ldb,
                    float* __restrict__ C, int ldc,
                    const float* __restrict__ bias,
                    int M, int Ktot, int Nw)
{
  __shared__ __bf16 As[BM * LDSK];
  __shared__ __bf16 Bs[BN * LDSK];
  const int tid  = threadIdx.x;
  const int lane = tid & 31;
  const int wv   = tid >> 5;
  const int m0   = blockIdx.x * BM;
  const int n0   = blockIdx.y * BN;

  v8f acc[4];
  #pragma unroll
  for (int t = 0; t < 4; ++t)
    #pragma unroll
    for (int i = 0; i < 8; ++i) acc[t][i] = 0.f;

  const int arow = tid >> 1;          // 0..127
  const int akh  = (tid & 1) * 16;    // K half within BK
  const int bn   = tid >> 2;          // 0..63
  const int bks  = (tid & 3) * 8;     // K chunk within BK
  const int kg   = lane >> 4;         // fragment K-group
  const int mr   = lane & 15;         // fragment row/col

  const int rowc = (m0 + arow < M) ? (m0 + arow) : (M - 1);  // clamp, no mask
  const float*  ap = A + (size_t)rowc * lda + akh;
  const __bf16* bp = Bt + (size_t)(n0 + bn) * ldb + bks;
  __bf16* adp = &As[arow * LDSK + akh];
  __bf16* bdp = &Bs[bn * LDSK + bks];

  for (int kb = 0; kb < Ktot; kb += BK) {
    { // A tile: fp32 -> bf16, 4x b128 loads, 2x b128 LDS stores
      const float* apk = ap + kb;
      float4 v0, v1, v2, v3;
      if (!KTAIL) {
        v0 = *(const float4*)(apk + 0);
        v1 = *(const float4*)(apk + 4);
        v2 = *(const float4*)(apk + 8);
        v3 = *(const float4*)(apk + 12);
      } else {
        const float4 z4 = make_float4(0.f, 0.f, 0.f, 0.f);
        int k = kb + akh;
        v0 = (k + 0  < Ktot) ? *(const float4*)(apk + 0)  : z4;
        v1 = (k + 4  < Ktot) ? *(const float4*)(apk + 4)  : z4;
        v2 = (k + 8  < Ktot) ? *(const float4*)(apk + 8)  : z4;
        v3 = (k + 12 < Ktot) ? *(const float4*)(apk + 12) : z4;
      }
      *(v8bf*)(adp)     = pack8(v0, v1);
      *(v8bf*)(adp + 8) = pack8(v2, v3);
      if (kb + BK < Ktot)                          // global_prefetch_b8
        __builtin_prefetch(apk + BK, 0, 3);
    }
    { // B tile: one unconditional 16B load + 16B LDS store per thread
      *(v8bf*)bdp = *(const v8bf*)(bp + kb);
    }
    __syncthreads();

    // A fragment 16x32: lane m=mr, elems K = {8kg..8kg+7} ++ {16+8kg..23+8kg}
    v16bf af;
    {
      const __bf16* rp = &As[(16 * wv + mr) * LDSK];
      v8bf a0 = *(const v8bf*)(rp + 8 * kg);
      v8bf a1 = *(const v8bf*)(rp + 16 + 8 * kg);
      #pragma unroll
      for (int i = 0; i < 8; ++i) { af[i] = a0[i]; af[i + 8] = a1[i]; }
    }
    // B fragment 32x16: lane n=mr, elems K = {8kg..8kg+7} ++ {16+8kg..23+8kg}
    #pragma unroll
    for (int t = 0; t < 4; ++t) {
      const __bf16* rp = &Bs[(16 * t + mr) * LDSK];
      v8bf b0 = *(const v8bf*)(rp + 8 * kg);
      v8bf b1 = *(const v8bf*)(rp + 16 + 8 * kg);
      v16bf bfv;
      #pragma unroll
      for (int i = 0; i < 8; ++i) { bfv[i] = b0[i]; bfv[i + 8] = b1[i]; }
      acc[t] = __builtin_amdgcn_wmma_f32_16x16x32_bf16(
          false, af, false, bfv, (short)0, acc[t], false, false);
    }
    __syncthreads();
  }

  // epilogue: D layout -> M = v + 8*kg (+16*wv), N = mr (+16*t)
  const int rbase = m0 + 16 * wv + 8 * kg;
  #pragma unroll
  for (int t = 0; t < 4; ++t) {
    int gn = n0 + 16 * t + mr;
    if (gn >= Nw) continue;
    float bv = bias ? bias[gn] : 0.f;
    #pragma unroll
    for (int v = 0; v < 8; ++v) {
      int gr = rbase + v;
      if (gr < M) C[(size_t)gr * ldc + gn] = acc[t][v] + bv;
    }
  }
}

// ---------------- attention scalars: el/er per (node,head); init emax/den ----
__global__ void attn_prep_kernel(const float* __restrict__ z,
                                 const float* __restrict__ al,
                                 const float* __restrict__ ar,
                                 float* el, float* er, float* emax, float* den,
                                 int Nn, int H) {
  int i = blockIdx.x * blockDim.x + threadIdx.x;
  if (i >= Nn * H) return;
  int n = i / H, h = i - n * H;
  const float* zr = z + (size_t)n * H * 64 + (size_t)h * 64;
  const float* a = al + h * 64;
  const float* b = ar + h * 64;
  float sl = 0.f, sr = 0.f;
  #pragma unroll 8
  for (int d = 0; d < 64; ++d) { float zv = zr[d]; sl += zv * a[d]; sr += zv * b[d]; }
  el[i] = sl; er[i] = sr; emax[i] = -1e30f; den[i] = 0.f;
}

__global__ void fill_kernel(float* p, long long n, float v) {
  long long i = (long long)blockIdx.x * blockDim.x + threadIdx.x;
  if (i < n) p[i] = v;
}

__device__ __forceinline__ void atomicMaxF(float* a, float v) {
  if (v >= 0.f) atomicMax((int*)a, __float_as_int(v));
  else atomicMin((unsigned int*)a, __float_as_uint(v));
}

__global__ void edge_max_kernel(const float* __restrict__ el, const float* __restrict__ er,
                                const int* __restrict__ src, const int* __restrict__ dst,
                                float* __restrict__ sc, float* __restrict__ emax,
                                int E, int H) {
  int e = blockIdx.x * blockDim.x + threadIdx.x;
  if (e >= E) return;
  int s = src[e], d = dst[e];
  for (int h = 0; h < H; ++h) {
    float x = el[s * H + h] + er[d * H + h];
    x = x > 0.f ? x : 0.2f * x;               // leaky relu, slope 0.2
    sc[(size_t)e * H + h] = x;
    atomicMaxF(&emax[d * H + h], x);
  }
}

__global__ void edge_exp_kernel(float* __restrict__ sc, const float* __restrict__ emax,
                                float* __restrict__ den, const int* __restrict__ dst,
                                int E, int H) {
  int e = blockIdx.x * blockDim.x + threadIdx.x;
  if (e >= E) return;
  int d = dst[e];
  for (int h = 0; h < H; ++h) {
    float x = __expf(sc[(size_t)e * H + h] - emax[d * H + h]);
    sc[(size_t)e * H + h] = x;
    atomicAdd(&den[d * H + h], x);
  }
}

__global__ void edge_agg_kernel(const float* __restrict__ sc, const float* __restrict__ den,
                                const float* __restrict__ z,
                                const int* __restrict__ src, const int* __restrict__ dst,
                                float* __restrict__ agg, int E, int H, int F) {
  long long i = (long long)blockIdx.x * blockDim.x + threadIdx.x;
  long long tot = (long long)E * H * 16;
  if (i >= tot) return;
  int c = (int)(i & 15);
  long long eh = i >> 4;                       // e*H + h
  int h = (int)(eh % H);
  int e = (int)(eh / H);
  int s = src[e], d = dst[e];
  float w = sc[eh] / den[d * H + h];
  const float4 zv = *(const float4*)(z + (size_t)s * F + h * 64 + c * 4);
  float* op = agg + (size_t)d * F + h * 64 + c * 4;
  atomicAdd(op + 0, w * zv.x);
  atomicAdd(op + 1, w * zv.y);
  atomicAdd(op + 2, w * zv.z);
  atomicAdd(op + 3, w * zv.w);
}

__global__ void post_kernel(const float* __restrict__ agg, const float* __restrict__ bias,
                            const float* __restrict__ res, float* __restrict__ out,
                            long long total, int F, int act) {
  long long i = (long long)blockIdx.x * blockDim.x + threadIdx.x;
  if (i >= total) return;
  int f = (int)(i % F);
  float v = agg[i] + bias[f];
  if (res) v += res[i];
  if (act) v = v > 0.f ? v : (__expf(v) - 1.f); // ELU
  out[i] = v;
}

// ---------------- host side ----------------
static inline void launch_gemm(hipStream_t s, const float* A, int lda, const __bf16* Bt,
                               int ldb, float* C, int ldc, const float* bias,
                               int M, int K, int Nw) {
  dim3 grid((M + BM - 1) / BM, (Nw + BN - 1) / BN);
  if (K % BK == 0)
    gemm_bf16_wmma<false><<<grid, 256, 0, s>>>(A, lda, Bt, ldb, C, ldc, bias, M, K, Nw);
  else
    gemm_bf16_wmma<true><<<grid, 256, 0, s>>>(A, lda, Bt, ldb, C, ldc, bias, M, K, Nw);
}

extern "C" void kernel_launch(void* const* d_in, const int* in_sizes, int n_in,
                              void* d_out, int out_size, void* d_ws, size_t ws_size,
                              hipStream_t stream)
{
  const float* f0 = (const float*)d_in[0];
  const float* f1 = (const float*)d_in[1];
  const float* f2 = (const float*)d_in[2];
  const float* nt0 = (const float*)d_in[3];
  const float* nt1 = (const float*)d_in[4];
  const float* nt2 = (const float*)d_in[5];
  const float* fcW0 = (const float*)d_in[6];  const float* fcb0 = (const float*)d_in[7];
  const float* fcW1 = (const float*)d_in[8];  const float* fcb1 = (const float*)d_in[9];
  const float* fcW2 = (const float*)d_in[10]; const float* fcb2 = (const float*)d_in[11];
  const float* ntW0 = (const float*)d_in[12]; const float* ntb0 = (const float*)d_in[13];
  const float* ntW1 = (const float*)d_in[14]; const float* ntb1 = (const float*)d_in[15];
  const float* ntW2 = (const float*)d_in[16]; const float* ntb2 = (const float*)d_in[17];
  const float* gW0 = (const float*)d_in[18];  const float* gal0 = (const float*)d_in[19];
  const float* gar0 = (const float*)d_in[20]; const float* gb0 = (const float*)d_in[21];
  const float* gW1 = (const float*)d_in[22];  const float* gal1 = (const float*)d_in[23];
  const float* gar1 = (const float*)d_in[24]; const float* gb1 = (const float*)d_in[25];
  const float* gW2 = (const float*)d_in[26];  const float* gal2 = (const float*)d_in[27];
  const float* gar2 = (const float*)d_in[28]; const float* gb2 = (const float*)d_in[29];
  const float* resW2 = (const float*)d_in[30];
  const float* linW = (const float*)d_in[31]; const float* linb = (const float*)d_in[32];
  const int* srcI = (const int*)d_in[33];
  const int* dstI = (const int*)d_in[34];
  (void)in_sizes; (void)n_in; (void)out_size; (void)ws_size;

  const int N = NNODES, E = NEDGES;

  // ---- workspace carve-up: bf16 weights, transposed+padded [Npad][Kpad] ----
  __bf16* wb = (__bf16*)d_ws;
  size_t wo = 0;
  auto wnext = [&](int Npad, int Kpad) { __bf16* p = wb + wo; wo += (size_t)Npad * Kpad; return p; };
  __bf16* fcW0t  = wnext(64, 256);
  __bf16* fcW1t  = wnext(64, 128);
  __bf16* fcW2t  = wnext(64, 64);
  __bf16* ntW0t  = wnext(64, 32);
  __bf16* ntW1t  = wnext(64, 32);
  __bf16* ntW2t  = wnext(64, 32);
  __bf16* gW0b   = wnext(256, 128);
  __bf16* gW1b   = wnext(256, 256);
  __bf16* gW2b   = wnext(64, 256);
  __bf16* resW2b = wnext(64, 256);
  __bf16* linWt  = wnext(64, 64);

  float* fp = (float*)((char*)d_ws + ((wo * sizeof(__bf16) + 255) & ~(size_t)255));
  size_t fo = 0;
  auto fnext = [&](size_t n) { float* p = fp + fo; fo += (n + 63) & ~(size_t)63; return p; };
  float* hcur = fnext((size_t)N * 256);
  float* zb   = fnext((size_t)N * 256);
  float* agg  = fnext((size_t)N * 256);
  float* res2 = fnext((size_t)N * 64);
  float* el   = fnext((size_t)N * 4);
  float* er   = fnext((size_t)N * 4);
  float* emax = fnext((size_t)N * 4);
  float* den  = fnext((size_t)N * 4);
  float* sc   = fnext((size_t)E * 4);

  // ---- weights -> bf16 [Npad][Kpad] (zero padded) ----
  auto wconv = [&](const float* W, __bf16* o, int K, int Nw, int Kpad, int Npad, int tr) {
    int tot = Kpad * Npad;
    wconv_kernel<<<(tot + 255) / 256, 256, 0, stream>>>(W, o, K, Nw, Kpad, Npad, tr);
  };
  wconv(fcW0, fcW0t, 256, 64, 256, 64, 0);   // [out,in] -> direct
  wconv(fcW1, fcW1t, 128, 64, 128, 64, 0);
  wconv(fcW2, fcW2t, 64, 64, 64, 64, 0);
  wconv(ntW0, ntW0t, 16, 64, 32, 64, 0);
  wconv(ntW1, ntW1t, 16, 64, 32, 64, 0);
  wconv(ntW2, ntW2t, 16, 64, 32, 64, 0);
  wconv(gW0, gW0b, 128, 256, 128, 256, 1);   // [in,out] -> transpose
  wconv(gW1, gW1b, 256, 256, 256, 256, 1);
  wconv(gW2, gW2b, 256, 64, 256, 64, 1);
  wconv(resW2, resW2b, 256, 64, 256, 64, 1);
  wconv(linW, linWt, 64, 16, 64, 64, 0);     // [out,in] 16x64 -> pad N to 64

  // ---- stage 0: per-type projections into hcur[N,128] (ld 256) ----
  launch_gemm(stream, f0, 256, fcW0t, 256, hcur, 256, fcb0, 20000, 256, 64);
  launch_gemm(stream, f1, 128, fcW1t, 128, hcur + (size_t)20000 * 256, 256, fcb1, 15000, 128, 64);
  launch_gemm(stream, f2,  64, fcW2t,  64, hcur + (size_t)35000 * 256, 256, fcb2, 15000, 64, 64);
  launch_gemm(stream, nt0, 16, ntW0t,  32, hcur + 64, 256, ntb0, 20000, 16, 64);
  launch_gemm(stream, nt1, 16, ntW1t,  32, hcur + (size_t)20000 * 256 + 64, 256, ntb1, 15000, 16, 64);
  launch_gemm(stream, nt2, 16, ntW2t,  32, hcur + (size_t)35000 * 256 + 64, 256, ntb2, 15000, 16, 64);

  float* hfinal = (float*)d_out + (size_t)N * 16;   // [N,64] part of output

  auto gat = [&](const __bf16* Bt, int ldb, int Fin, int H, const float* al, const float* ar,
                 const float* bias, const float* res, float* dest, int act) {
    int F = H * 64;
    launch_gemm(stream, hcur, 256, Bt, ldb, zb, F, nullptr, N, Fin, F);
    attn_prep_kernel<<<(N * H + 255) / 256, 256, 0, stream>>>(zb, al, ar, el, er, emax, den, N, H);
    long long nf = (long long)N * F;
    fill_kernel<<<(int)((nf + 255) / 256), 256, 0, stream>>>(agg, nf, 0.f);
    edge_max_kernel<<<(E + 255) / 256, 256, 0, stream>>>(el, er, srcI, dstI, sc, emax, E, H);
    edge_exp_kernel<<<(E + 255) / 256, 256, 0, stream>>>(sc, emax, den, dstI, E, H);
    long long tot = (long long)E * H * 16;
    edge_agg_kernel<<<(int)((tot + 255) / 256), 256, 0, stream>>>(sc, den, zb, srcI, dstI, agg, E, H, F);
    post_kernel<<<(int)((nf + 255) / 256), 256, 0, stream>>>(agg, bias, res, dest, nf, F, act);
  };

  // layer 0: no residual, ELU
  gat(gW0b, 128, 128, 4, gal0, gar0, gb0, nullptr, hcur, 1);
  // layer 1: identity residual, ELU (res == dest, elementwise safe)
  gat(gW1b, 256, 256, 4, gal1, gar1, gb1, hcur, hcur, 1);
  // layer 2: res_fc residual, no activation, 1 head -> hfinal
  launch_gemm(stream, hcur, 256, resW2b, 256, res2, 64, nullptr, N, 256, 64);
  gat(gW2b, 256, 256, 1, gal2, gar2, gb2, res2, hfinal, 0);
  // logits = hfinal @ linW.T + linb
  launch_gemm(stream, hfinal, 64, linWt, 64, (float*)d_out, 16, linb, N, 64, 16);
}
